// RWA_82875688944079
// MI455X (gfx1250) — matrix operand order repaired and checked by reading, not compiled, not run on hardware
//
#include <hip/hip_runtime.h>
#include <cstdint>
#include <cstddef>

// ---------------------------------------------------------------------------
// RWA recurrent network for MI455X (gfx1250, wave32, WMMA).
//
// Phase 1 (parallel): U/Gx/Ax = big bf16 WMMA GEMMs over all T*B rows.
// Phase 2 (sequential): 16 persistent workgroups (one per WGP), each owns 64
//   hidden columns; W_hh slices live in LDS (staged via async-to-LDS, 258KB),
//   n/d state in VGPRs; ht exchanged via double-buffered bf16 global buffer
//   (L2 resident) + atomic grid barrier with s_sleep backoff.
// ---------------------------------------------------------------------------

#define T_STEPS 1024
#define BATCH   128
#define DIN     512
#define DH      1024
#define NWG     16
#define COLS_PER_WG (DH / NWG)          // 64
#define LDS_ROW_HALFS (DH + 8)          // pad 16B per row -> avoid LDS bank conflicts

typedef __attribute__((ext_vector_type(16))) __bf16 v16bf;
typedef __attribute__((ext_vector_type(8)))  float  v8f;

union Frag16 { uint4 q[2]; v16bf v; };  // 32 bytes = one 16-bit WMMA operand per lane

// ---------------------------------------------------------------------------
// f32 -> bf16 bulk convert (8 elements / thread, b128 traffic both sides)
// ---------------------------------------------------------------------------
__global__ void k_f32_to_bf16(const float* __restrict__ src,
                              __bf16* __restrict__ dst, int n) {
  int i = (blockIdx.x * blockDim.x + threadIdx.x) * 8;
  if (i + 8 <= n) {
    float4 a = *(const float4*)(src + i);
    float4 b = *(const float4*)(src + i + 4);
    __bf16 o[8] = {(__bf16)a.x, (__bf16)a.y, (__bf16)a.z, (__bf16)a.w,
                   (__bf16)b.x, (__bf16)b.y, (__bf16)b.z, (__bf16)b.w};
    *(uint4*)(dst + i) = *(const uint4*)o;
  } else {
    for (; i < n; ++i) dst[i] = (__bf16)src[i];
  }
}

// Split [DH][DIN+DH] weight into bf16 x-part [DH][DIN] and h-part [DH][DH]
__global__ void k_split_w(const float* __restrict__ w,
                          __bf16* __restrict__ wx, __bf16* __restrict__ wh) {
  int j = blockIdx.x;
  const float* row = w + (size_t)j * (DIN + DH);
  for (int k = threadIdx.x; k < DIN; k += blockDim.x) wx[(size_t)j * DIN + k] = (__bf16)row[k];
  for (int k = threadIdx.x; k < DH;  k += blockDim.x) wh[(size_t)j * DH  + k] = (__bf16)row[DIN + k];
}

// ht0 = tanh(initial_state) broadcast over batch; also reset the grid barrier.
__global__ void k_init_state(const float* __restrict__ s0,
                             __bf16* __restrict__ ht0, unsigned* __restrict__ bar) {
  int i = blockIdx.x * blockDim.x + threadIdx.x;
  if (i == 0) *bar = 0u;
  if (i < BATCH * DH) ht0[i] = (__bf16)tanhf(s0[i % DH]);
}

// ---------------------------------------------------------------------------
// Phase 1: O[r, j] = sum_k Xb[r,k] * W[j,k] (+bias)  for  O in {U, Gx, Ax}
// grid = (T*B/128, DH/64, 3); block = 256 (8 waves). Each wave: 16 rows x 64 cols.
// K loop fully unrolled + A fragment software-pipelined so global-load latency
// overlaps the WMMA chain instead of producing wait(0) before every wmma.
// ---------------------------------------------------------------------------
__global__ __launch_bounds__(256) void k_pregemm(
    const __bf16* __restrict__ xb,                       // [T*B][DIN] bf16
    const __bf16* __restrict__ wu, const __bf16* __restrict__ wgx,
    const __bf16* __restrict__ wax,
    const float* __restrict__ u_b, const float* __restrict__ g_b,
    float* __restrict__ U, float* __restrict__ Gx, float* __restrict__ Ax) {
  const int lane = threadIdx.x & 31;
  const int wave = threadIdx.x >> 5;
  const int which = blockIdx.z;                          // 0:u 1:g 2:a
  const __bf16* W  = (which == 0) ? wu : (which == 1) ? wgx : wax;
  const float* bias = (which == 0) ? u_b : (which == 1) ? g_b : nullptr;
  float* O = (which == 0) ? U : (which == 1) ? Gx : Ax;

  const int row0 = blockIdx.x * 128 + wave * 16;
  const int col0 = blockIdx.y * 64;
  const int m    = lane & 15;
  const int hi   = lane >> 4;

  v8f acc[4] = {};
  const __bf16* arow = xb + (size_t)(row0 + m) * DIN;

  Frag16 aCur, aNxt;
  {
    const __bf16* ap = arow + hi * 8;
    aCur.q[0] = *(const uint4*)ap;
    aCur.q[1] = *(const uint4*)(ap + 16);
  }
#pragma unroll
  for (int kc = 0; kc < DIN; kc += 32) {
    if (kc + 32 < DIN) {                                 // prefetch next A frag
      const __bf16* ap = arow + (kc + 32) + hi * 8;
      aNxt.q[0] = *(const uint4*)ap;
      aNxt.q[1] = *(const uint4*)(ap + 16);
    }
    Frag16 b[4];                                         // batch all B loads first
#pragma unroll
    for (int ct = 0; ct < 4; ++ct) {
      const __bf16* bp = W + (size_t)(col0 + ct * 16 + m) * DIN + kc + hi * 8;
      b[ct].q[0] = *(const uint4*)bp;
      b[ct].q[1] = *(const uint4*)(bp + 16);
    }
#pragma unroll
    for (int ct = 0; ct < 4; ++ct)
      acc[ct] = __builtin_amdgcn_wmma_f32_16x16x32_bf16(
          false, aCur.v, false, b[ct].v, (short)0, acc[ct], false, false);
    aCur = aNxt;
  }
#pragma unroll
  for (int ct = 0; ct < 4; ++ct) {
    const int j = col0 + ct * 16 + m;
    const float bv = bias ? bias[j] : 0.0f;
#pragma unroll
    for (int v = 0; v < 8; ++v) {
      const int r = row0 + v + hi * 8;                   // C/D layout: M = v + hi*8
      O[(size_t)r * DH + j] = acc[ct][v] + bv;
    }
  }
}

// ---------------------------------------------------------------------------
// Phase 2: persistent recurrent kernel. 16 blocks x 256 threads; block c owns
// hidden columns [c*64, c*64+64). Weights resident in LDS, n/d in VGPRs.
// ---------------------------------------------------------------------------
extern __shared__ __bf16 s_lds[];

__global__ __launch_bounds__(256) void k_recurrent(
    const float* __restrict__ U, const float* __restrict__ Gx,
    const float* __restrict__ Ax,
    const __bf16* __restrict__ wgh, const __bf16* __restrict__ wah,
    __bf16* __restrict__ ht0, __bf16* __restrict__ ht1,
    float* __restrict__ out, unsigned* __restrict__ bar) {
  const int lane = threadIdx.x & 31;
  const int wave = threadIdx.x >> 5;
  const int col0 = blockIdx.x * COLS_PER_WG;
  const int m    = lane & 15;
  const int hi   = lane >> 4;

  __bf16* sWg = s_lds;                                   // LDS byte offset 0
  __bf16* sWa = s_lds + 64 * LDS_ROW_HALFS;              // LDS byte offset 64*2064

  // Stage the two 64x1024 bf16 W_hh slices into LDS with async-to-LDS DMA
  // (GLOBAL_LOAD_ASYNC_TO_LDS_B128, tracked by ASYNCcnt): no VGPR round-trip.
  // VDST operand carries the per-lane LDS byte offset (workgroup LDS base = 0,
  // dynamic LDS only). 16B per lane per issue; padded row stride = 2064B.
  {
    const int chunksPerMat = 64 * (DH / 8);              // 8192 x 16B chunks
    for (int idx = threadIdx.x; idx < chunksPerMat; idx += blockDim.x) {
      const int r = idx >> 7;                            // row within 64
      const int p = idx & 127;                           // 16B chunk within row
      const int ldsOffG = r * (LDS_ROW_HALFS * 2) + p * 16;
      const int ldsOffA = ldsOffG + 64 * LDS_ROW_HALFS * 2;
      const __bf16* gp = wgh + (size_t)(col0 + r) * DH + p * 8;
      const __bf16* ap = wah + (size_t)(col0 + r) * DH + p * 8;
      asm volatile("global_load_async_to_lds_b128 %0, %1, off"
                   :: "v"(ldsOffG), "v"(gp) : "memory");
      asm volatile("global_load_async_to_lds_b128 %0, %1, off"
                   :: "v"(ldsOffA), "v"(ap) : "memory");
    }
    asm volatile("s_wait_asynccnt 0x0" ::: "memory");
  }
  __syncthreads();

  v8f nacc[4] = {}, dacc[4] = {};                        // per-(row,col) RWA state
  const int aRow = wave * 16 + m;                        // batch row for A fragment

  for (int t = 0; t < T_STEPS; ++t) {
    const __bf16* htCur = (t & 1) ? ht1 : ht0;
    __bf16*       htNxt = (t & 1) ? ht0 : ht1;

    // Prefetch next step's activations into GL2 while we do WMMA work.
    if (t + 1 < T_STEPS) {
      const size_t p = ((size_t)(t + 1) * BATCH + aRow) * DH + col0 + lane;
      __builtin_prefetch(&U[p], 0, 1);
      __builtin_prefetch(&Gx[p], 0, 1);
      __builtin_prefetch(&Ax[p], 0, 1);
    }

    v8f g[4] = {}, a[4] = {};
    const __bf16* arow = htCur + (size_t)aRow * DH;

    Frag16 af, afN;
    {
      const __bf16* ap = arow + hi * 8;
      af.q[0] = *(const uint4*)ap;
      af.q[1] = *(const uint4*)(ap + 16);
    }
#pragma unroll 4
    for (int kc = 0; kc < DH; kc += 32) {
      if (kc + 32 < DH) {                                // pipeline next A frag
        const __bf16* ap = arow + (kc + 32) + hi * 8;
        afN.q[0] = *(const uint4*)ap;
        afN.q[1] = *(const uint4*)(ap + 16);
      }
      const int boff0 = m * LDS_ROW_HALFS + kc + hi * 8;
      Frag16 bg[4];
#pragma unroll
      for (int ct = 0; ct < 4; ++ct) {                   // batch 8 ds_load_b128
        const int boff = boff0 + ct * 16 * LDS_ROW_HALFS;
        bg[ct].q[0] = *(const uint4*)(sWg + boff);
        bg[ct].q[1] = *(const uint4*)(sWg + boff + 16);
      }
#pragma unroll
      for (int ct = 0; ct < 4; ++ct)
        g[ct] = __builtin_amdgcn_wmma_f32_16x16x32_bf16(
            false, af.v, false, bg[ct].v, (short)0, g[ct], false, false);
      Frag16 ba[4];
#pragma unroll
      for (int ct = 0; ct < 4; ++ct) {
        const int boff = boff0 + ct * 16 * LDS_ROW_HALFS;
        ba[ct].q[0] = *(const uint4*)(sWa + boff);
        ba[ct].q[1] = *(const uint4*)(sWa + boff + 16);
      }
#pragma unroll
      for (int ct = 0; ct < 4; ++ct)
        a[ct] = __builtin_amdgcn_wmma_f32_16x16x32_bf16(
            false, af.v, false, ba[ct].v, (short)0, a[ct], false, false);
      af = afN;
    }

    // Epilogue: u*tanh(g)*e^a accumulation, h = tanh(n/d), emit h.
#pragma unroll
    for (int ct = 0; ct < 4; ++ct) {
      const int j = col0 + ct * 16 + m;
#pragma unroll
      for (int v = 0; v < 8; ++v) {
        const int b = wave * 16 + v + hi * 8;
        const size_t idx = ((size_t)t * BATCH + b) * DH + j;   // == [t][b][j]
        const float uu = U[idx];
        const float gg = Gx[idx] + g[ct][v];
        const float aa = Ax[idx] + a[ct][v];
        const float ea = __expf(aa);
        nacc[ct][v] += uu * tanhf(gg) * ea;
        dacc[ct][v] += ea;
        const float h = tanhf(nacc[ct][v] / dacc[ct][v]);
        out[idx] = h;
        htNxt[(size_t)b * DH + j] = (__bf16)h;
        if (t == T_STEPS - 1)
          out[(size_t)T_STEPS * BATCH * DH + (size_t)b * DH + j] = h;
      }
    }

    // Device-wide step barrier (monotonic counter; reset each launch by k_init_state).
    __threadfence();
    __syncthreads();
    if (threadIdx.x == 0) {
      atomicAdd(bar, 1u);
      const unsigned target = (unsigned)(t + 1) * NWG;
      while (__hip_atomic_load(bar, __ATOMIC_ACQUIRE, __HIP_MEMORY_SCOPE_AGENT) < target)
        __builtin_amdgcn_s_sleep(1);
    }
    __syncthreads();
  }
}

// ---------------------------------------------------------------------------
extern "C" void kernel_launch(void* const* d_in, const int* in_sizes, int n_in,
                              void* d_out, int out_size, void* d_ws, size_t ws_size,
                              hipStream_t stream) {
  (void)in_sizes; (void)n_in; (void)out_size; (void)ws_size;
  const float* x   = (const float*)d_in[0];
  const float* s0  = (const float*)d_in[1];
  const float* u_W = (const float*)d_in[2];
  const float* u_b = (const float*)d_in[3];
  const float* g_W = (const float*)d_in[4];
  const float* g_b = (const float*)d_in[5];
  const float* a_W = (const float*)d_in[6];
  float* out = (float*)d_out;

  // Workspace carve (256B aligned).
  char* w = (char*)d_ws;
  size_t off = 0;
  auto take = [&](size_t bytes) {
    void* p = w + off;
    off += (bytes + 255) & ~(size_t)255;
    return p;
  };
  unsigned* bar = (unsigned*)take(sizeof(unsigned));
  __bf16* ht0 = (__bf16*)take((size_t)BATCH * DH * 2);
  __bf16* ht1 = (__bf16*)take((size_t)BATCH * DH * 2);
  __bf16* xb  = (__bf16*)take((size_t)T_STEPS * BATCH * DIN * 2);
  __bf16* wu  = (__bf16*)take((size_t)DH * DIN * 2);
  __bf16* wgx = (__bf16*)take((size_t)DH * DIN * 2);
  __bf16* wax = (__bf16*)take((size_t)DH * DIN * 2);
  __bf16* wgh = (__bf16*)take((size_t)DH * DH * 2);
  __bf16* wah = (__bf16*)take((size_t)DH * DH * 2);
  float*  U   = (float*)take((size_t)T_STEPS * BATCH * DH * 4);
  float*  Gx  = (float*)take((size_t)T_STEPS * BATCH * DH * 4);
  float*  Ax  = (float*)take((size_t)T_STEPS * BATCH * DH * 4);

  // Conversions + init.
  {
    const int n = T_STEPS * BATCH * DIN;
    k_f32_to_bf16<<<(n / 8 + 255) / 256, 256, 0, stream>>>(x, xb, n);
  }
  k_f32_to_bf16<<<(DH * DIN / 8 + 255) / 256, 256, 0, stream>>>(u_W, wu, DH * DIN);
  k_split_w<<<DH, 256, 0, stream>>>(g_W, wgx, wgh);
  k_split_w<<<DH, 256, 0, stream>>>(a_W, wax, wah);
  k_init_state<<<(BATCH * DH + 255) / 256, 256, 0, stream>>>(s0, ht0, bar);

  // Phase 1: parallel pre-GEMMs.
  k_pregemm<<<dim3(T_STEPS * BATCH / 128, DH / 64, 3), 256, 0, stream>>>(
      xb, wu, wgx, wax, u_b, g_b, U, Gx, Ax);

  // Phase 2: persistent recurrent kernel (one block per WGP; 258KB LDS each).
  const size_t ldsBytes = (size_t)2 * 64 * LDS_ROW_HALFS * sizeof(__bf16);
  static bool attrSet = [] {
    return hipFuncSetAttribute((const void*)k_recurrent,
                               hipFuncAttributeMaxDynamicSharedMemorySize,
                               (int)((size_t)2 * 64 * LDS_ROW_HALFS * 2)) == hipSuccess;
  }();
  (void)attrSet;
  k_recurrent<<<NWG, 256, ldsBytes, stream>>>(U, Gx, Ax, wgh, wah, ht0, ht1, out, bar);
}
